// SocialMobModel_73847667687686
// MI455X (gfx1250) — compile-verified, matrix operand order checked by reading, not run on hardware
//
#include <hip/hip_runtime.h>
#include <hip/hip_bf16.h>
#include <math.h>

#define BB 2
#define NN 1024
#define DD 256
#define HH 8
#define HDIM 32
#define FFD 1024
#define NDEPTH 2
#define RELHID 64
#define ATT_SCALE 0.1767766952966369f   // 32^-0.5

typedef __bf16 bf16;
typedef bf16  v16bf __attribute__((ext_vector_type(16)));
typedef bf16  v8bf  __attribute__((ext_vector_type(8)));
typedef float v8f   __attribute__((ext_vector_type(8)));

// Load a 16-element bf16 A/B fragment as two contiguous 8xbf16 (16B) chunks.
__device__ __forceinline__ v16bf ld16v(const bf16* p0, const bf16* p1) {
  v8bf lo = *(const v8bf*)p0;
  v8bf hi = *(const v8bf*)p1;
  v16bf r;
#pragma unroll
  for (int i = 0; i < 8; ++i) { r[i] = lo[i]; r[i + 8] = hi[i]; }
  return r;
}

__device__ __forceinline__ float gelu_exact(float x) {
  return 0.5f * x * (1.0f + erff(x * 0.7071067811865476f));
}

// CDNA5 async global->LDS copy: 16B per lane, tracked by ASYNCcnt.
__device__ __forceinline__ void async_copy_b128(void* lds_dst, const void* gsrc) {
  const unsigned int la = (unsigned int)(unsigned long long)lds_dst; // aperture low bits == LDS offset
  const unsigned long long ga = (unsigned long long)gsrc;
  asm volatile("global_load_async_to_lds_b128 %0, %1, off"
               :: "v"(la), "v"(ga) : "memory");
}
__device__ __forceinline__ void wait_asynccnt0() {
  asm volatile("s_wait_asynccnt 0" ::: "memory");
}

// ---------------- f32 -> bf16 weight conversion ----------------
__global__ void cvt_bf16_kernel(const float* __restrict__ in, bf16* __restrict__ out, int n) {
  int i = blockIdx.x * blockDim.x + threadIdx.x;
  if (i < n) out[i] = (bf16)in[i];
}

// ---------------- LayerNorm over D=256, one block per token ----------------
__global__ __launch_bounds__(256) void ln_kernel(const float* __restrict__ x,
                                                 const float* __restrict__ g,
                                                 const float* __restrict__ b,
                                                 bf16* __restrict__ out) {
  __shared__ float red[256];
  const int row = blockIdx.x;
  const int c = threadIdx.x;
  const float v = x[(size_t)row * DD + c];
  red[c] = v;
  __syncthreads();
  for (int s = 128; s > 0; s >>= 1) { if (c < s) red[c] += red[c + s]; __syncthreads(); }
  const float mean = red[0] * (1.0f / DD);
  __syncthreads();
  const float d = v - mean;
  red[c] = d * d;
  __syncthreads();
  for (int s = 128; s > 0; s >>= 1) { if (c < s) red[c] += red[c + s]; __syncthreads(); }
  const float var = red[0] * (1.0f / DD);
  out[(size_t)row * DD + c] = (bf16)(d * rsqrtf(var + 1e-5f) * g[c] + b[c]);
}

// ---------------- generic bf16 WMMA GEMM: out = act(A @ W + bias) (+res) ---
// A: [M,K] bf16 row-major. W: [K,Nc] bf16 row-major.
// Block tile 64x64, 8 waves (4x2); each wave owns a 16x32 output (2 WMMAs/K-step
// sharing one A fragment). A tile staged via async global->LDS (ASYNCcnt).
__global__ __launch_bounds__(256) void wmma_gemm_kernel(
    const bf16* __restrict__ A, const bf16* __restrict__ W,
    const float* __restrict__ bias, const float* __restrict__ resid,
    float* __restrict__ outf, bf16* __restrict__ outb,
    int M, int K, int Nc, int do_gelu) {
  __shared__ __align__(16) bf16 As[64][40];   // 64x32 A tile, padded rows (80B)
  __shared__ __align__(16) bf16 Bt[64][40];   // 32x64 W tile, stored transposed [n][k]
  const int tid  = threadIdx.x;
  const int lane = tid & 31;
  const int w    = tid >> 5;
  const int wm   = w >> 1, wn = w & 1;
  const int row0 = blockIdx.x * 64;
  const int col0 = blockIdx.y * 64;

  const int ar = (tid * 8) >> 5, ac = (tid * 8) & 31;  // A staging coords (64x32)
  const int bk = (tid * 8) >> 6, bn = (tid * 8) & 63;  // W staging coords (32x64)
  const int fr  = lane & 15;                           // fragment row / col
  const int kk0 = (lane < 16) ? 0 : 8;                 // ISA 16-bit A/B K-phase

  v8f acc0 = {}, acc1 = {};
  const int nK = K >> 5;
  for (int kt = 0; kt < nK; ++kt) {
    const int k0 = kt << 5;
    // stage A tile: async global->LDS, 16B per lane
    async_copy_b128(&As[ar][ac], A + (size_t)(row0 + ar) * K + (k0 + ac));
    // stage W tile transposed so B-fragment reads are contiguous in K
    const bf16* wp = W + (size_t)(k0 + bk) * Nc + (col0 + bn);
    v8bf wv = *(const v8bf*)wp;
#pragma unroll
    for (int i = 0; i < 8; ++i) Bt[bn + i][bk] = wv[i];
    if (kt + 1 < nK) {
      __builtin_prefetch(A + (size_t)(row0 + ar) * K + (k0 + 32 + ac), 0, 1);
      __builtin_prefetch(W + (size_t)(k0 + 32 + bk) * Nc + (col0 + bn), 0, 1);
    }
    wait_asynccnt0();
    __syncthreads();
    v16bf afrag = ld16v(&As[wm * 16 + fr][kk0], &As[wm * 16 + fr][kk0 + 16]);
    v16bf b0 = ld16v(&Bt[wn * 32 + fr][kk0],      &Bt[wn * 32 + fr][kk0 + 16]);
    v16bf b1 = ld16v(&Bt[wn * 32 + 16 + fr][kk0], &Bt[wn * 32 + 16 + fr][kk0 + 16]);
    acc0 = __builtin_amdgcn_wmma_f32_16x16x32_bf16(false, afrag, false, b0,
                                                   (short)0, acc0, false, false);
    acc1 = __builtin_amdgcn_wmma_f32_16x16x32_bf16(false, afrag, false, b1,
                                                   (short)0, acc1, false, false);
    __syncthreads();
  }
  // epilogue: C layout -> element (M = r + 8*(lane>=16), N = lane&15)
  const int m_off = (lane >= 16) ? 8 : 0;
  const int colA = col0 + wn * 32 + fr;
  const int colB = colA + 16;
  const float biasA = bias ? bias[colA] : 0.0f;
  const float biasB = bias ? bias[colB] : 0.0f;
#pragma unroll
  for (int r = 0; r < 8; ++r) {
    const int row = row0 + wm * 16 + r + m_off;
    float va = acc0[r] + biasA;
    float vb = acc1[r] + biasB;
    if (do_gelu) { va = gelu_exact(va); vb = gelu_exact(vb); }
    if (resid) {
      va += resid[(size_t)row * Nc + colA];
      vb += resid[(size_t)row * Nc + colB];
    }
    if (outf) {
      outf[(size_t)row * Nc + colA] = va;
      outf[(size_t)row * Nc + colB] = vb;
    }
    if (outb) {
      outb[(size_t)row * Nc + colA] = (bf16)va;
      outb[(size_t)row * Nc + colB] = (bf16)vb;
    }
  }
}

// ---------------- rope + split qkv -> bf16 [B,H,N,HD] ----------------
__global__ __launch_bounds__(256) void rope_pack_kernel(const float* __restrict__ qkv,
                                                        bf16* __restrict__ qo,
                                                        bf16* __restrict__ ko,
                                                        bf16* __restrict__ vo) {
  const int t  = blockIdx.x * blockDim.x + threadIdx.x;  // 0..B*H*N-1
  const int bh = t >> 10;
  const int n  = t & 1023;
  const int b  = bh >> 3, h = bh & 7;
  const float* base = qkv + ((size_t)(b * NN + n)) * (3 * HH * HDIM) + h * HDIM;
  const size_t ob = (size_t)t * HDIM;
#pragma unroll
  for (int i = 0; i < 16; ++i) {
    const float invf = __expf(-(float)i * (9.210340371976184f / 16.0f)); // 10000^(-i/16)
    float s, c;
    __sincosf((float)n * invf, &s, &c);
    const float q1 = base[i],       q2 = base[i + 16];
    const float k1 = base[256 + i], k2 = base[256 + i + 16];
    qo[ob + i]      = (bf16)(q1 * c - q2 * s);
    qo[ob + 16 + i] = (bf16)(q2 * c + q1 * s);
    ko[ob + i]      = (bf16)(k1 * c - k2 * s);
    ko[ob + 16 + i] = (bf16)(k2 * c + k1 * s);
    vo[ob + i]      = (bf16)base[512 + i];
    vo[ob + 16 + i] = (bf16)base[512 + i + 16];
  }
}

// ---------------- pairwise rel-bias MLP -> rel[b,h,i,j] f32 ----------------
__global__ __launch_bounds__(256) void rel_bias_kernel(
    const float* __restrict__ coords, const float* __restrict__ vels,
    const float* __restrict__ W1, const float* __restrict__ b1,
    const float* __restrict__ W2, const float* __restrict__ b2,
    float* __restrict__ rel) {
  __shared__ float sW1[5 * RELHID];
  __shared__ float sb1[RELHID];
  __shared__ float sW2[RELHID * HH];
  __shared__ float sb2[HH];
  const int t = threadIdx.x;
  if (t < 320) sW1[t] = W1[t];
  if (t < 64)  sb1[t] = b1[t];
  sW2[t] = W2[t]; sW2[t + 256] = W2[t + 256];
  if (t < 8)   sb2[t] = b2[t];
  __syncthreads();

  const int gid = blockIdx.x * 256 + t;          // 0 .. B*N*N-1
  const int b = gid >> 20;
  const int rem = gid & (NN * NN - 1);
  const int i = rem >> 10, j = rem & 1023;
  const float f0 = coords[(size_t)(b * NN + i) * 2 + 0] - coords[(size_t)(b * NN + j) * 2 + 0];
  const float f1 = coords[(size_t)(b * NN + i) * 2 + 1] - coords[(size_t)(b * NN + j) * 2 + 1];
  const float f2 = sqrtf(f0 * f0 + f1 * f1);
  const float f3 = vels[(size_t)(b * NN + i) * 2 + 0] - vels[(size_t)(b * NN + j) * 2 + 0];
  const float f4 = vels[(size_t)(b * NN + i) * 2 + 1] - vels[(size_t)(b * NN + j) * 2 + 1];

  float acc[HH];
#pragma unroll
  for (int h = 0; h < HH; ++h) acc[h] = sb2[h];
#pragma unroll 4
  for (int r = 0; r < RELHID; ++r) {
    float hs = f0 * sW1[r] + f1 * sW1[64 + r] + f2 * sW1[128 + r] +
               f3 * sW1[192 + r] + f4 * sW1[256 + r] + sb1[r];
    const float ge = gelu_exact(hs);
    const float* w2 = &sW2[r * HH];
#pragma unroll
    for (int h = 0; h < HH; ++h) acc[h] += ge * w2[h];
  }
#pragma unroll
  for (int h = 0; h < HH; ++h)
    rel[(((size_t)(b * HH + h)) * NN + i) * NN + j] = acc[h];
}

// ---------------- flash attention, one 16-row q-tile per wave ----------------
__global__ __launch_bounds__(128) void attn_kernel(
    const bf16* __restrict__ qg, const bf16* __restrict__ kg,
    const bf16* __restrict__ vg, const float* __restrict__ rel,
    bf16* __restrict__ out) {
  __shared__ __align__(16) bf16 Pls[4][16][40];  // per-wave P repack slab
  const int lane = threadIdx.x & 31;
  const int w    = threadIdx.x >> 5;
  const int tile = blockIdx.x * 4 + w;           // 0..B*H*(N/16)-1
  const int i0   = (tile & 63) << 4;
  const int bh   = tile >> 6;
  const int b    = bh >> 3, h = bh & 7;

  const int fr    = lane & 15;
  const int kk0   = (lane < 16) ? 0 : 8;
  const int m_off = (lane >= 16) ? 8 : 0;

  const bf16* qptr = qg + (size_t)bh * NN * HDIM;
  const bf16* kptr = kg + (size_t)bh * NN * HDIM;
  const bf16* vptr = vg + (size_t)bh * NN * HDIM;
  const float* relp = rel + (size_t)bh * NN * NN;

  const v16bf qa = ld16v(&qptr[(i0 + fr) * HDIM + kk0], &qptr[(i0 + fr) * HDIM + kk0 + 16]);

  v8f o0 = {}, o1 = {};
  float mrow[8], lrow[8];
#pragma unroll
  for (int r = 0; r < 8; ++r) { mrow[r] = -INFINITY; lrow[r] = 0.0f; }

  for (int j0 = 0; j0 < NN; j0 += 32) {
    // K^T B-fragments: column n = key j0+n, K dim = HD (contiguous in memory)
    v16bf kb0 = ld16v(&kptr[(j0 + fr) * HDIM + kk0],      &kptr[(j0 + fr) * HDIM + kk0 + 16]);
    v16bf kb1 = ld16v(&kptr[(j0 + 16 + fr) * HDIM + kk0], &kptr[(j0 + 16 + fr) * HDIM + kk0 + 16]);
    v8f z = {};
    v8f s0 = __builtin_amdgcn_wmma_f32_16x16x32_bf16(false, qa, false, kb0, (short)0, z, false, false);
    v8f s1 = __builtin_amdgcn_wmma_f32_16x16x32_bf16(false, qa, false, kb1, (short)0, z, false, false);
#pragma unroll
    for (int r = 0; r < 8; ++r) {
      const size_t mi = (size_t)(i0 + r + m_off) * NN;
      float a = s0[r] * ATT_SCALE + relp[mi + j0 + fr];
      float c = s1[r] * ATT_SCALE + relp[mi + j0 + 16 + fr];
      float mx = fmaxf(a, c);
#pragma unroll
      for (int d = 8; d >= 1; d >>= 1) mx = fmaxf(mx, __shfl_xor(mx, d, 32)); // stays in 16-lane half
      const float mnew  = fmaxf(mrow[r], mx);
      const float alpha = __expf(mrow[r] - mnew);
      const float p0 = __expf(a - mnew);
      const float p1 = __expf(c - mnew);
      float rs = p0 + p1;
#pragma unroll
      for (int d = 8; d >= 1; d >>= 1) rs += __shfl_xor(rs, d, 32);
      lrow[r] = lrow[r] * alpha + rs;
      mrow[r] = mnew;
      o0[r] *= alpha; o1[r] *= alpha;
      Pls[w][r + m_off][fr]      = (bf16)p0;   // C-layout -> LDS row-major
      Pls[w][r + m_off][fr + 16] = (bf16)p1;
    }
    asm volatile("s_wait_dscnt 0" ::: "memory");
    v16bf pa = ld16v(&Pls[w][fr][kk0], &Pls[w][fr][kk0 + 16]);
    // V B-fragments (gather: K dim = key index, strided by HD)
    v16bf vb0, vb1;
#pragma unroll
    for (int i = 0; i < 8; ++i) {
      vb0[i]     = vptr[(j0 + kk0 + i) * HDIM + fr];
      vb0[i + 8] = vptr[(j0 + kk0 + 16 + i) * HDIM + fr];
      vb1[i]     = vptr[(j0 + kk0 + i) * HDIM + 16 + fr];
      vb1[i + 8] = vptr[(j0 + kk0 + 16 + i) * HDIM + 16 + fr];
    }
    o0 = __builtin_amdgcn_wmma_f32_16x16x32_bf16(false, pa, false, vb0, (short)0, o0, false, false);
    o1 = __builtin_amdgcn_wmma_f32_16x16x32_bf16(false, pa, false, vb1, (short)0, o1, false, false);
  }
#pragma unroll
  for (int r = 0; r < 8; ++r) {
    const float inv = (lrow[r] > 0.0f) ? 1.0f / lrow[r] : 0.0f;  // nan_to_num
    const size_t row = (size_t)b * NN + i0 + r + m_off;
    out[row * (HH * HDIM) + h * HDIM + fr]      = (bf16)(o0[r] * inv);
    out[row * (HH * HDIM) + h * HDIM + 16 + fr] = (bf16)(o1[r] * inv);
  }
}

// ---------------- host orchestration ----------------
extern "C" void kernel_launch(void* const* d_in, const int* in_sizes, int n_in,
                              void* d_out, int out_size, void* d_ws, size_t ws_size,
                              hipStream_t stream) {
  const float* x      = (const float*)d_in[0];
  const float* coords = (const float*)d_in[1];
  const float* vels   = (const float*)d_in[2];
  const float* ln1g   = (const float*)d_in[3];
  const float* ln1b   = (const float*)d_in[4];
  const float* Wqkv   = (const float*)d_in[5];
  const float* rW1    = (const float*)d_in[6];
  const float* rb1    = (const float*)d_in[7];
  const float* rW2    = (const float*)d_in[8];
  const float* rb2    = (const float*)d_in[9];
  const float* Wout   = (const float*)d_in[10];
  const float* bout   = (const float*)d_in[11];
  const float* ln2g   = (const float*)d_in[12];
  const float* ln2b   = (const float*)d_in[13];
  const float* Wf1    = (const float*)d_in[14];
  const float* bf1p   = (const float*)d_in[15];
  const float* Wf2    = (const float*)d_in[16];
  const float* bf2p   = (const float*)d_in[17];

  const int M = BB * NN;  // 2048 tokens
  char* ws = (char*)d_ws;
  size_t off = 0;
  auto carve = [&](size_t bytes) -> void* {
    void* p = ws + off;
    off += (bytes + 255) & ~(size_t)255;
    return p;
  };
  float* xf    = (float*)carve((size_t)M * DD * 4);
  bf16*  hbf   = (bf16*)carve((size_t)M * DD * 2);
  float* qkvf  = (float*)carve((size_t)M * 3 * HH * HDIM * 4);
  bf16*  qbf   = (bf16*)carve((size_t)BB * HH * NN * HDIM * 2);
  bf16*  kbf   = (bf16*)carve((size_t)BB * HH * NN * HDIM * 2);
  bf16*  vbf   = (bf16*)carve((size_t)BB * HH * NN * HDIM * 2);
  float* relb  = (float*)carve((size_t)BB * HH * NN * NN * 4);
  bf16*  atbf  = (bf16*)carve((size_t)M * DD * 2);
  bf16*  ffbf  = (bf16*)carve((size_t)M * FFD * 2);
  bf16*  Wqkvb = (bf16*)carve((size_t)NDEPTH * DD * 3 * HH * HDIM * 2);
  bf16*  Woutb = (bf16*)carve((size_t)NDEPTH * DD * DD * 2);
  bf16*  Wf1b  = (bf16*)carve((size_t)NDEPTH * DD * FFD * 2);
  bf16*  Wf2b  = (bf16*)carve((size_t)NDEPTH * FFD * DD * 2);

  hipMemcpyAsync(xf, x, (size_t)M * DD * 4, hipMemcpyDeviceToDevice, stream);
  auto cvt = [&](const float* src, bf16* dst, int n) {
    cvt_bf16_kernel<<<dim3((n + 255) / 256), dim3(256), 0, stream>>>(src, dst, n);
  };
  cvt(Wqkv, Wqkvb, NDEPTH * DD * 3 * HH * HDIM);
  cvt(Wout, Woutb, NDEPTH * DD * DD);
  cvt(Wf1,  Wf1b,  NDEPTH * DD * FFD);
  cvt(Wf2,  Wf2b,  NDEPTH * FFD * DD);

  for (int l = 0; l < NDEPTH; ++l) {
    ln_kernel<<<dim3(M), dim3(256), 0, stream>>>(xf, ln1g + l * DD, ln1b + l * DD, hbf);
    wmma_gemm_kernel<<<dim3(M / 64, (3 * HH * HDIM) / 64), dim3(256), 0, stream>>>(
        hbf, Wqkvb + (size_t)l * DD * 3 * HH * HDIM, nullptr, nullptr,
        qkvf, nullptr, M, DD, 3 * HH * HDIM, 0);
    rope_pack_kernel<<<dim3((BB * HH * NN) / 256), dim3(256), 0, stream>>>(qkvf, qbf, kbf, vbf);
    rel_bias_kernel<<<dim3((BB * NN * NN) / 256), dim3(256), 0, stream>>>(
        coords, vels, rW1 + l * 5 * RELHID, rb1 + l * RELHID,
        rW2 + l * RELHID * HH, rb2 + l * HH, relb);
    attn_kernel<<<dim3((BB * HH * (NN / 16)) / 4), dim3(128), 0, stream>>>(
        qbf, kbf, vbf, relb, atbf);
    wmma_gemm_kernel<<<dim3(M / 64, DD / 64), dim3(256), 0, stream>>>(
        atbf, Woutb + (size_t)l * DD * DD, bout + l * DD, xf, xf, nullptr, M, DD, DD, 0);
    ln_kernel<<<dim3(M), dim3(256), 0, stream>>>(xf, ln2g + l * DD, ln2b + l * DD, hbf);
    wmma_gemm_kernel<<<dim3(M / 64, FFD / 64), dim3(256), 0, stream>>>(
        hbf, Wf1b + (size_t)l * DD * FFD, bf1p + l * FFD, nullptr, nullptr, ffbf, M, DD, FFD, 1);
    wmma_gemm_kernel<<<dim3(M / 64, DD / 64), dim3(256), 0, stream>>>(
        ffbf, Wf2b + (size_t)l * FFD * DD, bf2p + l * DD, xf, xf, nullptr, M, FFD, DD, 0);
  }
  hipMemcpyAsync(d_out, xf, (size_t)M * DD * 4, hipMemcpyDeviceToDevice, stream);
}
